// FNetLayer_75797582840223
// MI455X (gfx1250) — matrix operand order repaired and checked by reading, not compiled, not run on hardware
//
#include <hip/hip_runtime.h>
#include <hip/hip_bf16.h>

// ---------------------------------------------------------------------------
// FNet layer for MI455X (gfx1250, wave32).
//   x_ft = Re(FFT_S(FFT_H(x)))          -> butterfly FFTs in LDS (fp32)
//   h    = LN(x + x_ft)                 -> fused, also emits bf16 copy of h
//   ffn  = gelu(h@w1+b1)@w2+b2          -> bf16 WMMA (v_wmma_f32_16x16x32_bf16)
//   out  = LN(h + ffn)
// Workspace layout (368 MB):
//   [0,64M)    h_f32        [64M,96M)  h_bf16     [96M,104M) w1_bf16
//   [104M,112M)w2_bf16      [112M,176M) fft_re -> later ffn_f32
//   [176M,240M)fft_im       [240M,368M) x_ft (64M) -> later act_bf16 (128M)
// ---------------------------------------------------------------------------

#if __has_builtin(__builtin_amdgcn_global_load_async_to_lds_b128)
#define HAVE_ASYNC_LDS 1
#endif

typedef __attribute__((ext_vector_type(16))) __bf16 v16bf;
typedef __attribute__((ext_vector_type(8)))  float  v8f;
typedef __attribute__((ext_vector_type(4)))  int    v4i;
#if defined(HAVE_ASYNC_LDS)
typedef __attribute__((address_space(1))) v4i v4i_g;   // global (device) int4
typedef __attribute__((address_space(3))) v4i v4i_l;   // LDS int4
#endif

union Frag16 { uint4 q[2]; v16bf v; };

static __device__ __forceinline__ unsigned short f2bf(float f) {
    unsigned int u = __float_as_uint(f);
    u += 0x7FFFu + ((u >> 16) & 1u);        // round to nearest even
    return (unsigned short)(u >> 16);
}

// ------------------------------- fp32 -> bf16 ------------------------------
__global__ void cvt_bf16_kernel(const float* __restrict__ in,
                                unsigned short* __restrict__ out, int n) {
    int i = blockIdx.x * 256 + threadIdx.x;
    if (i < n) out[i] = f2bf(in[i]);
}

// ------------------------------- FFT helpers -------------------------------
template<int NFFT>
static __device__ __forceinline__ void fft_stages(float* sr, float* si, int t) {
    int sh = 0;
    for (int len = 2; len <= NFFT; len <<= 1, ++sh) {
        const int half = len >> 1;
        const float ang0 = -6.28318530717958647692f / (float)len;
        for (int bi = t; bi < (NFFT >> 1); bi += 256) {
            int pos = bi & (half - 1);
            int i0  = ((bi >> sh) << (sh + 1)) | pos;
            int i1  = i0 + half;
            float sv, cv;
            __sincosf(ang0 * (float)pos, &sv, &cv);
            float er = sr[i1], ei = si[i1];
            float tr = er * cv - ei * sv;
            float ti = er * sv + ei * cv;
            float ur = sr[i0], ui = si[i0];
            sr[i0] = ur + tr;  si[i0] = ui + ti;
            sr[i1] = ur - tr;  si[i1] = ui - ti;
        }
        __syncthreads();
    }
}

// 1024-point FFT along hidden dim; real input, complex output. 1 row / block.
__global__ __launch_bounds__(256) void fft_h_kernel(const float* __restrict__ x,
                                                    float* __restrict__ ore,
                                                    float* __restrict__ oim) {
    __shared__ float sr[1024], si[1024];
    const int t = threadIdx.x;
    const size_t base = (size_t)blockIdx.x * 1024;
    for (int i = t; i < 1024; i += 256) {
        int j = __brev((unsigned)i) >> 22;      // 10-bit reversal
        sr[j] = x[base + i];
        si[j] = 0.f;
    }
    __syncthreads();
    fft_stages<1024>(sr, si, t);
    for (int i = t; i < 1024; i += 256) {
        ore[base + i] = sr[i];
        oim[base + i] = si[i];
    }
}

// 4096-point FFT along seq dim; complex input, real part out. 1 column/block.
__global__ __launch_bounds__(256) void fft_s_kernel(const float* __restrict__ re,
                                                    const float* __restrict__ im,
                                                    float* __restrict__ xft) {
    __shared__ float sr[4096], si[4096];        // 32 KB of 320 KB LDS
    const int t = threadIdx.x;
    const int col = blockIdx.x;                 // 0..4095 : (b, h)
    const size_t base = (size_t)(col >> 10) * (4096u * 1024u) + (col & 1023);
    for (int i = t; i < 4096; i += 256) {
        int j = __brev((unsigned)i) >> 20;      // 12-bit reversal
        size_t g = base + (size_t)i * 1024;
        sr[j] = re[g];
        si[j] = im[g];
    }
    __syncthreads();
    fft_stages<4096>(sr, si, t);
    for (int i = t; i < 4096; i += 256)
        xft[base + (size_t)i * 1024] = sr[i];
}

// ------------------------ LayerNorm(a+b) over H=1024 -----------------------
__global__ __launch_bounds__(256) void ln_kernel(const float* __restrict__ a,
                                                 const float* __restrict__ b,
                                                 const float* __restrict__ gamma,
                                                 const float* __restrict__ beta,
                                                 float* __restrict__ out_f32,
                                                 unsigned short* __restrict__ out_bf16) {
    __shared__ float red[256];
    const int t = threadIdx.x;
    const size_t base = (size_t)blockIdx.x * 1024;
    float v[4]; float s = 0.f;
    for (int i = 0; i < 4; ++i) {
        int c = t + i * 256;
        v[i] = a[base + c] + b[base + c];
        s += v[i];
    }
    red[t] = s; __syncthreads();
    for (int off = 128; off > 0; off >>= 1) {
        if (t < off) red[t] += red[t + off];
        __syncthreads();
    }
    const float mean = red[0] * (1.f / 1024.f);
    __syncthreads();
    float s2 = 0.f;
    for (int i = 0; i < 4; ++i) { float d = v[i] - mean; s2 += d * d; }
    red[t] = s2; __syncthreads();
    for (int off = 128; off > 0; off >>= 1) {
        if (t < off) red[t] += red[t + off];
        __syncthreads();
    }
    const float rstd = rsqrtf(red[0] * (1.f / 1024.f) + 1e-5f);
    for (int i = 0; i < 4; ++i) {
        int c = t + i * 256;
        float o = (v[i] - mean) * rstd * gamma[c] + beta[c];
        if (out_f32)  out_f32[base + c]  = o;
        if (out_bf16) out_bf16[base + c] = f2bf(o);
    }
}

// ----------------------- bf16 WMMA GEMM: C = A*B + bias --------------------
// A: [M,K] bf16 row-major, B: [K,N] bf16 row-major.
// Block tile 128x128, K-chunk 64, 8 waves (2x4), each wave 4x2 16x16 subtiles.
// LDS rows padded to 72 elems (144 B): 16B-aligned rows, conflict-free frag
// reads (9*m mod 16 distinct). amdgpu_waves_per_eu(4) -> 256-VGPR budget so
// the 64 accumulator VGPRs stay resident through the epilogue (no spill).
// A tile is staged with global_load_async_to_lds_b128 (ASYNCcnt) when the
// builtin exists; B tile is repacked through VGPRs (needs [k][n]->[n][k]).
#define TSTR 72
template<bool DO_GELU, bool OUT_BF16>
__global__ __attribute__((amdgpu_waves_per_eu(4))) __launch_bounds__(256)
void gemm_bf16_kernel(
    const unsigned short* __restrict__ A, const unsigned short* __restrict__ Bm,
    const float* __restrict__ bias, void* __restrict__ out, int N, int K) {
    __shared__ __align__(16) unsigned short sA[128 * TSTR];   // 18 KB
    __shared__ __align__(16) unsigned short sB[128 * TSTR];   // 18 KB

    const int t = threadIdx.x;
    const int lane = t & 31, wave = t >> 5;
    const int wm = wave >> 2, wn = wave & 3;            // 2 x 4 wave grid
    const int m0 = blockIdx.x * 128, n0 = blockIdx.y * 128;
    const int khalf = lane >> 4, ln16 = lane & 15;

    v8f acc[4][2] = {};

    for (int k0 = 0; k0 < K; k0 += 64) {
        // --- stage A tile 128x64: async global->LDS DMA if available ---
#if defined(HAVE_ASYNC_LDS)
        for (int i = 0; i < 4; ++i) {
            int l = t + i * 256;                        // uint4 id 0..1023
            int row = l >> 3, c4 = l & 7;
            const uint4* g = (const uint4*)(A + (size_t)(m0 + row) * K + k0) + c4;
            __builtin_amdgcn_global_load_async_to_lds_b128(
                (v4i_g*)(void*)g,
                (v4i_l*)(sA + row * TSTR + c4 * 8),
                0, 0);
        }
#else
        {
            uint4 aw[4];
            for (int i = 0; i < 4; ++i) {
                int l = t + i * 256;
                int row = l >> 3, c4 = l & 7;
                aw[i] = *((const uint4*)(A + (size_t)(m0 + row) * K + k0) + c4);
            }
            for (int i = 0; i < 4; ++i) {
                int l = t + i * 256;
                int row = l >> 3, c4 = l & 7;
                *(uint4*)(sA + row * TSTR + c4 * 8) = aw[i];
            }
        }
#endif
        // --- stage B tile 64x128, transposed to [n][k]; pack 4 K-rows ---
        {
            unsigned int bw[4][4];
            for (int i = 0; i < 4; ++i) {
                int l = t + i * 256;                    // 0..1023
                int np = l & 63;                        // n-pair 0..63
                int k  = (l >> 6) * 4;                  // 0,4,...,60
                const unsigned short* gp = Bm + (size_t)(k0 + k) * N + n0 + np * 2;
                bw[i][0] = *(const unsigned int*)(gp);
                bw[i][1] = *(const unsigned int*)(gp + N);
                bw[i][2] = *(const unsigned int*)(gp + 2 * (size_t)N);
                bw[i][3] = *(const unsigned int*)(gp + 3 * (size_t)N);
            }
            for (int i = 0; i < 4; ++i) {
                int l = t + i * 256;
                int np = l & 63;
                int k  = (l >> 6) * 4;
                uint2 lo, hi;
                lo.x = (bw[i][0] & 0xFFFFu) | (bw[i][1] << 16);     // (k..k+1, n0)
                lo.y = (bw[i][2] & 0xFFFFu) | (bw[i][3] << 16);     // (k+2..k+3, n0)
                hi.x = (bw[i][0] >> 16) | (bw[i][1] & 0xFFFF0000u); // (k..k+1, n1)
                hi.y = (bw[i][2] >> 16) | (bw[i][3] & 0xFFFF0000u);
                *(uint2*)(sB + (np * 2 + 0) * TSTR + k) = lo;
                *(uint2*)(sB + (np * 2 + 1) * TSTR + k) = hi;
            }
        }
        // --- prefetch next K-chunk into cache (global_prefetch_b8) ---
        if (k0 + 64 < K) {
            if (t < 128) {
                __builtin_prefetch(A + (size_t)(m0 + t) * K + (k0 + 64), 0, 1);
            } else {
                int r = t - 128;
                __builtin_prefetch(Bm + (size_t)(k0 + 64 + (r >> 1)) * N + n0 + (r & 1) * 64, 0, 1);
            }
        }
#if defined(HAVE_ASYNC_LDS)
        asm volatile("s_wait_asynccnt 0x0" ::: "memory");
#endif
        __syncthreads();

        for (int ks = 0; ks < 64; ks += 32) {
            Frag16 af[4], bfr[2];
            for (int ms = 0; ms < 4; ++ms) {
                const unsigned short* p =
                    sA + (wm * 64 + ms * 16 + ln16) * TSTR + ks + khalf * 8;
                af[ms].q[0] = *(const uint4*)(p);        // K = ks+khalf*8 ..
                af[ms].q[1] = *(const uint4*)(p + 16);   // K = ks+16+khalf*8 ..
            }
            for (int ns = 0; ns < 2; ++ns) {
                const unsigned short* p =
                    sB + (wn * 32 + ns * 16 + ln16) * TSTR + ks + khalf * 8;
                bfr[ns].q[0] = *(const uint4*)(p);       // K = ks+khalf*8 ..
                bfr[ns].q[1] = *(const uint4*)(p + 16);  // K = ks+16+khalf*8 ..
            }
            for (int ms = 0; ms < 4; ++ms)
                for (int ns = 0; ns < 2; ++ns)
                    acc[ms][ns] = __builtin_amdgcn_wmma_f32_16x16x32_bf16(
                        false, af[ms].v, false, bfr[ns].v,
                        (short)0, acc[ms][ns], false, false);
        }
        __syncthreads();
    }

    // --- epilogue: bias (+ exact GELU) ---
    for (int ms = 0; ms < 4; ++ms)
        for (int ns = 0; ns < 2; ++ns) {
            const int n = n0 + wn * 32 + ns * 16 + ln16;
            const float bv = bias[n];
            for (int v = 0; v < 8; ++v) {
                const int m = m0 + wm * 64 + ms * 16 + v + khalf * 8;
                float f = acc[ms][ns][v] + bv;
                if (DO_GELU) f = 0.5f * f * (1.f + erff(f * 0.70710678118654752f));
                if (OUT_BF16)
                    ((unsigned short*)out)[(size_t)m * N + n] = f2bf(f);
                else
                    ((float*)out)[(size_t)m * N + n] = f;
            }
        }
}

// ---------------------------------------------------------------------------
extern "C" void kernel_launch(void* const* d_in, const int* in_sizes, int n_in,
                              void* d_out, int out_size, void* d_ws, size_t ws_size,
                              hipStream_t stream) {
    const float* x   = (const float*)d_in[0];
    const float* w1  = (const float*)d_in[1];
    const float* b1  = (const float*)d_in[2];
    const float* w2  = (const float*)d_in[3];
    const float* b2  = (const float*)d_in[4];
    const float* g1  = (const float*)d_in[5];
    const float* be1 = (const float*)d_in[6];
    const float* g2  = (const float*)d_in[7];
    const float* be2 = (const float*)d_in[8];

    char* ws = (char*)d_ws;
    float*          h_f32  = (float*)(ws);
    unsigned short* h_bf16 = (unsigned short*)(ws + (64ull  << 20));
    unsigned short* w1b    = (unsigned short*)(ws + (96ull  << 20));
    unsigned short* w2b    = (unsigned short*)(ws + (104ull << 20));
    float*          fre    = (float*)(ws + (112ull << 20));   // later ffn
    float*          fim    = (float*)(ws + (176ull << 20));
    float*          xft    = (float*)(ws + (240ull << 20));   // later act
    unsigned short* act    = (unsigned short*)(ws + (240ull << 20));
    float*          ffn    = fre;

    // weights -> bf16 (4M elems each)
    cvt_bf16_kernel<<<16384, 256, 0, stream>>>(w1, w1b, 1024 * 4096);
    cvt_bf16_kernel<<<16384, 256, 0, stream>>>(w2, w2b, 4096 * 1024);

    // 2D FFT, real part
    fft_h_kernel<<<16384, 256, 0, stream>>>(x, fre, fim);
    fft_s_kernel<<<4096,  256, 0, stream>>>(fre, fim, xft);

    // h = LN(x + x_ft)   (f32 + bf16)
    ln_kernel<<<16384, 256, 0, stream>>>(x, xft, g1, be1, h_f32, h_bf16);

    // act = gelu(h @ w1 + b1)  : M=16384, N=4096, K=1024
    gemm_bf16_kernel<true, true><<<dim3(128, 32), 256, 0, stream>>>(
        h_bf16, w1b, b1, (void*)act, 4096, 1024);

    // ffn = act @ w2 + b2      : M=16384, N=1024, K=4096
    gemm_bf16_kernel<false, false><<<dim3(128, 8), 256, 0, stream>>>(
        act, w2b, b2, (void*)ffn, 1024, 4096);

    // out = LN(h + ffn)
    ln_kernel<<<16384, 256, 0, stream>>>(h_f32, ffn, g2, be2, (float*)d_out, nullptr);
}
